// Transformer_6674379178719
// MI455X (gfx1250) — compile-verified, matrix operand order checked by reading, not run on hardware
//
#include <hip/hip_runtime.h>
#include <hip/hip_bf16.h>

// ---------------------------------------------------------------------------
// 6-layer pre-norm transformer encoder forward, bf16 WMMA (gfx1250 / MI455X)
// ---------------------------------------------------------------------------

typedef __attribute__((ext_vector_type(16))) __bf16 v16bf;
typedef __attribute__((ext_vector_type(8)))  __bf16 v8bf;
typedef __attribute__((ext_vector_type(8)))  float  v8f;

#define D_MODEL 1024
#define D_FF    2048
#define N_LAYERS 6
#define BATCH   4
#define SEQ     1024
#define NHEADS  16
#define DK      64
#define ROWS    (BATCH * SEQ) /* 4096 */

#define SHUF16(a, b) __builtin_shufflevector((a), (b), 0,1,2,3,4,5,6,7,8,9,10,11,12,13,14,15)
#define WMMA_BF16(a, b, c) \
  __builtin_amdgcn_wmma_f32_16x16x32_bf16(false, (a), false, (b), (short)0, (c), false, false)

__device__ __forceinline__ __bf16 f2bf(float f) {
  unsigned u = __builtin_bit_cast(unsigned, f);
  u += 0x7FFFu + ((u >> 16) & 1u);               // round-to-nearest-even
  unsigned short h = (unsigned short)(u >> 16);
  return __builtin_bit_cast(__bf16, h);
}

// ---------------- elementwise f32 -> bf16 ----------------
__global__ void cvt_f32_bf16(const float* __restrict__ in, __bf16* __restrict__ out, int n) {
  int i = blockIdx.x * 256 + threadIdx.x;
  if (i < n) out[i] = f2bf(in[i]);
}

// ---------------- LayerNorm (torch: unbiased std, /(std+eps)) ----------------
__global__ void layernorm_k(const float* __restrict__ x,
                            const float* __restrict__ gamma,
                            const float* __restrict__ beta,
                            __bf16* __restrict__ outB,
                            float* __restrict__ outF) {
  const int row = blockIdx.x;                    // ROWS blocks
  const float* xr = x + (size_t)row * D_MODEL;
  float v[4]; float s = 0.f, ss = 0.f;
#pragma unroll
  for (int i = 0; i < 4; ++i) {
    v[i] = xr[threadIdx.x + 256 * i];
    s += v[i]; ss += v[i] * v[i];
  }
#pragma unroll
  for (int m = 16; m >= 1; m >>= 1) {
    s  += __shfl_xor(s,  m, 32);
    ss += __shfl_xor(ss, m, 32);
  }
  __shared__ float r0[8], r1[8];
  const int w = threadIdx.x >> 5;
  if ((threadIdx.x & 31) == 0) { r0[w] = s; r1[w] = ss; }
  __syncthreads();
  s = 0.f; ss = 0.f;
#pragma unroll
  for (int i = 0; i < 8; ++i) { s += r0[i]; ss += r1[i]; }
  const float mean = s * (1.0f / D_MODEL);
  float var = (ss - (float)D_MODEL * mean * mean) * (1.0f / (D_MODEL - 1));
  var = var < 0.f ? 0.f : var;
  const float inv = 1.0f / (sqrtf(var) + 1e-6f);
#pragma unroll
  for (int i = 0; i < 4; ++i) {
    const int c = threadIdx.x + 256 * i;
    const float y = gamma[c] * (v[i] - mean) * inv + beta[c];
    if (outB) outB[(size_t)row * D_MODEL + c] = f2bf(y);
    else      outF[(size_t)row * D_MODEL + c] = y;
  }
}

// ---------------- GEMM: C[M,N] = A[M,K] * W[N,K]^T (+bias, relu, residual) --
// Wave: 16x64 output tile. Block 256 = 8 waves -> 128x64. Grid (N/64, M/128).
__global__ void __launch_bounds__(256)
gemm_wmma(const __bf16* __restrict__ A, const __bf16* __restrict__ W,
          const float* __restrict__ bias, const float* __restrict__ resid,
          float* __restrict__ outF, __bf16* __restrict__ outB,
          int M, int N, int K, int relu) {
  const int lane = threadIdx.x & 31;
  const int wave = threadIdx.x >> 5;
  const int lo = lane & 15;
  const int hi = lane >> 4;
  const int m0 = blockIdx.y * 128 + wave * 16;
  const int n0 = blockIdx.x * 64;

  const __bf16* Ap  = A + (size_t)(m0 + lo) * K + 8 * hi;
  const __bf16* Wp0 = W + (size_t)(n0 +  0 + lo) * K + 8 * hi;
  const __bf16* Wp1 = W + (size_t)(n0 + 16 + lo) * K + 8 * hi;
  const __bf16* Wp2 = W + (size_t)(n0 + 32 + lo) * K + 8 * hi;
  const __bf16* Wp3 = W + (size_t)(n0 + 48 + lo) * K + 8 * hi;

  v8f c0 = {}, c1 = {}, c2 = {}, c3 = {};
  for (int kk = 0; kk < K; kk += 32) {
    const v16bf a = SHUF16(*(const v8bf*)(Ap + kk), *(const v8bf*)(Ap + kk + 16));
    v16bf b;
    b = SHUF16(*(const v8bf*)(Wp0 + kk), *(const v8bf*)(Wp0 + kk + 16));
    c0 = WMMA_BF16(a, b, c0);
    b = SHUF16(*(const v8bf*)(Wp1 + kk), *(const v8bf*)(Wp1 + kk + 16));
    c1 = WMMA_BF16(a, b, c1);
    b = SHUF16(*(const v8bf*)(Wp2 + kk), *(const v8bf*)(Wp2 + kk + 16));
    c2 = WMMA_BF16(a, b, c2);
    b = SHUF16(*(const v8bf*)(Wp3 + kk), *(const v8bf*)(Wp3 + kk + 16));
    c3 = WMMA_BF16(a, b, c3);
  }
#pragma unroll
  for (int r = 0; r < 8; ++r) {
    const int row = m0 + r + 8 * hi;            // C layout: lanes16-31 hold M=r+8
    const float vr[4] = { c0[r], c1[r], c2[r], c3[r] };
#pragma unroll
    for (int j = 0; j < 4; ++j) {
      const int col = n0 + 16 * j + lo;
      float val = vr[j] + bias[col];
      if (relu) val = val > 0.f ? val : 0.f;
      const size_t idx = (size_t)row * N + col;
      if (resid) val += resid[idx];
      if (outF) outF[idx] = val;
      if (outB) outB[idx] = f2bf(val);
    }
  }
}

// ---------------- V transpose: [b,s,h*64+c] -> [b,h,c,s] ----------------
__global__ void vtrans_k(const __bf16* __restrict__ v, __bf16* __restrict__ vt) {
  const int idx = blockIdx.x * 256 + threadIdx.x;   // 4*16*64*1024 total
  const int s = idx & 1023;
  const int c = (idx >> 10) & 63;
  const int h = (idx >> 16) & 15;
  const int b = idx >> 20;
  vt[idx] = v[(size_t)(b * SEQ + s) * D_MODEL + h * DK + c];
}

// ---------------- Flash attention, one wave per (b, h, 16-row Q tile) -------
__global__ void __launch_bounds__(256)
attn_wmma(const __bf16* __restrict__ Q, const __bf16* __restrict__ Km,
          const __bf16* __restrict__ Vt, const int* __restrict__ mask,
          __bf16* __restrict__ O) {
  __shared__ __align__(16) __bf16 plds[8 * 16 * 32];  // per-wave P tile (16x32)
  const int lane = threadIdx.x & 31;
  const int wave = threadIdx.x >> 5;
  const int lo = lane & 15, hi = lane >> 4;
  const int task = blockIdx.x * 8 + wave;             // 4096 tasks
  const int qt = task & 63;
  const int h  = (task >> 6) & 15;
  const int b  = task >> 10;
  const int q0 = qt * 16;

  // Q fragments over dk=64 (two K=32 steps), resident for the whole loop
  const __bf16* qp = Q + (size_t)(b * SEQ + q0 + lo) * D_MODEL + h * DK + 8 * hi;
  const v16bf qa0 = SHUF16(*(const v8bf*)(qp),      *(const v8bf*)(qp + 16));
  const v16bf qa1 = SHUF16(*(const v8bf*)(qp + 32), *(const v8bf*)(qp + 48));

  const int* mrow = mask + ((size_t)b * SEQ + q0) * SEQ;

  v8f o0 = {}, o1 = {}, o2 = {}, o3 = {};
  float mrun[8], lrun[8];
#pragma unroll
  for (int r = 0; r < 8; ++r) { mrun[r] = -3.0e38f; lrun[r] = 0.f; }

  __bf16* pw = plds + wave * 512;

  for (int kv = 0; kv < SEQ; kv += 32) {
    // ---- scores: two 16x16 tiles over 32 KV rows
    const __bf16* kp0 = Km + (size_t)(b * SEQ + kv + lo)      * D_MODEL + h * DK + 8 * hi;
    const __bf16* kp1 = Km + (size_t)(b * SEQ + kv + 16 + lo) * D_MODEL + h * DK + 8 * hi;
    v8f s0 = {}, s1 = {};
    v16bf kb;
    kb = SHUF16(*(const v8bf*)(kp0),      *(const v8bf*)(kp0 + 16)); s0 = WMMA_BF16(qa0, kb, s0);
    kb = SHUF16(*(const v8bf*)(kp0 + 32), *(const v8bf*)(kp0 + 48)); s0 = WMMA_BF16(qa1, kb, s0);
    kb = SHUF16(*(const v8bf*)(kp1),      *(const v8bf*)(kp1 + 16)); s1 = WMMA_BF16(qa0, kb, s1);
    kb = SHUF16(*(const v8bf*)(kp1 + 32), *(const v8bf*)(kp1 + 48)); s1 = WMMA_BF16(qa1, kb, s1);

    // ---- scale + mask + online softmax (rows live in 16-lane halves)
    float p0[8], p1[8];
#pragma unroll
    for (int r = 0; r < 8; ++r) {
      const int row = r + 8 * hi;
      float a0v = s0[r] * 0.125f;                 // 1/sqrt(64)
      float a1v = s1[r] * 0.125f;
      if (mrow[(size_t)row * SEQ + kv + lo] == 0)      a0v = -1e9f;
      if (mrow[(size_t)row * SEQ + kv + 16 + lo] == 0) a1v = -1e9f;
      p0[r] = a0v; p1[r] = a1v;
    }
#pragma unroll
    for (int r = 0; r < 8; ++r) {
      float vmax = p0[r] > p1[r] ? p0[r] : p1[r];
#pragma unroll
      for (int mI = 8; mI >= 1; mI >>= 1) {
        const float o = __shfl_xor(vmax, mI, 32);
        vmax = vmax > o ? vmax : o;
      }
      const float mnew  = mrun[r] > vmax ? mrun[r] : vmax;
      const float alpha = __expf(mrun[r] - mnew);
      const float e0 = __expf(p0[r] - mnew);
      const float e1 = __expf(p1[r] - mnew);
      float psum = e0 + e1;
#pragma unroll
      for (int mI = 8; mI >= 1; mI >>= 1) psum += __shfl_xor(psum, mI, 32);
      lrun[r] = lrun[r] * alpha + psum;
      mrun[r] = mnew;
      o0[r] *= alpha; o1[r] *= alpha; o2[r] *= alpha; o3[r] *= alpha;
      const int row = r + 8 * hi;                 // C layout -> LDS (row-major 16x32)
      pw[row * 32 + lo]      = f2bf(e0);
      pw[row * 32 + 16 + lo] = f2bf(e1);
    }
    // ---- P (16x32) as A-fragment from LDS (wave-local; LDS ops in-order)
    const v16bf pa = SHUF16(*(const v8bf*)(pw + lo * 32 + 8 * hi),
                            *(const v8bf*)(pw + lo * 32 + 16 + 8 * hi));
    // ---- O += P * V : Vt rows are KV-contiguous
    const __bf16* vp = Vt + (size_t)((b * NHEADS + h) * DK + lo) * SEQ + kv + 8 * hi;
    v16bf vb;
    vb = SHUF16(*(const v8bf*)(vp),                *(const v8bf*)(vp + 16));            o0 = WMMA_BF16(pa, vb, o0);
    vb = SHUF16(*(const v8bf*)(vp + 16 * SEQ),     *(const v8bf*)(vp + 16 * SEQ + 16)); o1 = WMMA_BF16(pa, vb, o1);
    vb = SHUF16(*(const v8bf*)(vp + 32 * SEQ),     *(const v8bf*)(vp + 32 * SEQ + 16)); o2 = WMMA_BF16(pa, vb, o2);
    vb = SHUF16(*(const v8bf*)(vp + 48 * SEQ),     *(const v8bf*)(vp + 48 * SEQ + 16)); o3 = WMMA_BF16(pa, vb, o3);
  }
#pragma unroll
  for (int r = 0; r < 8; ++r) {
    const int row = r + 8 * hi;
    const float inv = 1.0f / lrun[r];
    const size_t base = (size_t)(b * SEQ + q0 + row) * D_MODEL + h * DK + lo;
    O[base]      = f2bf(o0[r] * inv);
    O[base + 16] = f2bf(o1[r] * inv);
    O[base + 32] = f2bf(o2[r] * inv);
    O[base + 48] = f2bf(o3[r] * inv);
  }
}

// ---------------------------------------------------------------------------
extern "C" void kernel_launch(void* const* d_in, const int* in_sizes, int n_in,
                              void* d_out, int out_size, void* d_ws, size_t ws_size,
                              hipStream_t stream) {
  (void)in_sizes; (void)n_in; (void)out_size; (void)ws_size;
  const float* word = (const float*)d_in[0];
  const int*   mask = (const int*)  d_in[1];
  const float* wq = (const float*)d_in[2];  const float* bq = (const float*)d_in[3];
  const float* wk = (const float*)d_in[4];  const float* bk = (const float*)d_in[5];
  const float* wv = (const float*)d_in[6];  const float* bv = (const float*)d_in[7];
  const float* wo = (const float*)d_in[8];  const float* bo = (const float*)d_in[9];
  const float* w1 = (const float*)d_in[10]; const float* b1 = (const float*)d_in[11];
  const float* w2 = (const float*)d_in[12]; const float* b2 = (const float*)d_in[13];
  const float* ln1a = (const float*)d_in[14]; const float* ln1b = (const float*)d_in[15];
  const float* ln2a = (const float*)d_in[16]; const float* ln2b = (const float*)d_in[17];
  const float* lnfa = (const float*)d_in[18]; const float* lnfb = (const float*)d_in[19];

  // ---- workspace layout (bytes, 256-aligned) ----
  char* ws = (char*)d_ws;
  const size_t SZ_ATTN_W = (size_t)N_LAYERS * D_MODEL * D_MODEL * 2;  // 12.6 MB
  const size_t SZ_FFN_W  = (size_t)N_LAYERS * D_FF * D_MODEL * 2;     // 25.2 MB
  size_t off = 0;
  __bf16* wq_bf = (__bf16*)(ws + off); off += SZ_ATTN_W;
  __bf16* wk_bf = (__bf16*)(ws + off); off += SZ_ATTN_W;
  __bf16* wv_bf = (__bf16*)(ws + off); off += SZ_ATTN_W;
  __bf16* wo_bf = (__bf16*)(ws + off); off += SZ_ATTN_W;
  __bf16* w1_bf = (__bf16*)(ws + off); off += SZ_FFN_W;
  __bf16* w2_bf = (__bf16*)(ws + off); off += SZ_FFN_W;
  float*  x     = (float*) (ws + off); off += (size_t)ROWS * D_MODEL * 4;
  __bf16* xn_bf = (__bf16*)(ws + off); off += (size_t)ROWS * D_MODEL * 2;
  __bf16* q_bf  = (__bf16*)(ws + off); off += (size_t)ROWS * D_MODEL * 2;
  __bf16* k_bf  = (__bf16*)(ws + off); off += (size_t)ROWS * D_MODEL * 2;
  __bf16* v_bf  = (__bf16*)(ws + off); off += (size_t)ROWS * D_MODEL * 2;
  __bf16* vt_bf = (__bf16*)(ws + off); off += (size_t)ROWS * D_MODEL * 2;
  __bf16* at_bf = (__bf16*)(ws + off); off += (size_t)ROWS * D_MODEL * 2;
  __bf16* h_bf  = (__bf16*)(ws + off); off += (size_t)ROWS * D_FF * 2;

  // ---- convert weights to bf16 ----
  const int nA = N_LAYERS * D_MODEL * D_MODEL;
  const int nF = N_LAYERS * D_FF * D_MODEL;
  cvt_f32_bf16<<<(nA + 255) / 256, 256, 0, stream>>>(wq, wq_bf, nA);
  cvt_f32_bf16<<<(nA + 255) / 256, 256, 0, stream>>>(wk, wk_bf, nA);
  cvt_f32_bf16<<<(nA + 255) / 256, 256, 0, stream>>>(wv, wv_bf, nA);
  cvt_f32_bf16<<<(nA + 255) / 256, 256, 0, stream>>>(wo, wo_bf, nA);
  cvt_f32_bf16<<<(nF + 255) / 256, 256, 0, stream>>>(w1, w1_bf, nF);
  cvt_f32_bf16<<<(nF + 255) / 256, 256, 0, stream>>>(w2, w2_bf, nF);

  hipMemcpyAsync(x, word, (size_t)ROWS * D_MODEL * 4, hipMemcpyDeviceToDevice, stream);

  const dim3 gD(D_MODEL / 64, ROWS / 128);   // (16, 32)
  const dim3 gF(D_FF / 64,    ROWS / 128);   // (32, 32)

  for (int i = 0; i < N_LAYERS; ++i) {
    const size_t oA = (size_t)i * D_MODEL * D_MODEL;
    const size_t oW = (size_t)i * D_FF * D_MODEL;
    // --- attention sublayer ---
    layernorm_k<<<ROWS, 256, 0, stream>>>(x, ln1a + i * D_MODEL, ln1b + i * D_MODEL, xn_bf, nullptr);
    gemm_wmma<<<gD, 256, 0, stream>>>(xn_bf, wq_bf + oA, bq + i * D_MODEL, nullptr, nullptr, q_bf,
                                      ROWS, D_MODEL, D_MODEL, 0);
    gemm_wmma<<<gD, 256, 0, stream>>>(xn_bf, wk_bf + oA, bk + i * D_MODEL, nullptr, nullptr, k_bf,
                                      ROWS, D_MODEL, D_MODEL, 0);
    gemm_wmma<<<gD, 256, 0, stream>>>(xn_bf, wv_bf + oA, bv + i * D_MODEL, nullptr, nullptr, v_bf,
                                      ROWS, D_MODEL, D_MODEL, 0);
    vtrans_k<<<(ROWS * D_MODEL) / 256, 256, 0, stream>>>(v_bf, vt_bf);
    attn_wmma<<<512, 256, 0, stream>>>(q_bf, k_bf, vt_bf, mask, at_bf);
    gemm_wmma<<<gD, 256, 0, stream>>>(at_bf, wo_bf + oA, bo + i * D_MODEL, x, x, nullptr,
                                      ROWS, D_MODEL, D_MODEL, 0);   // x += O-proj
    // --- FFN sublayer ---
    layernorm_k<<<ROWS, 256, 0, stream>>>(x, ln2a + i * D_MODEL, ln2b + i * D_MODEL, xn_bf, nullptr);
    gemm_wmma<<<gF, 256, 0, stream>>>(xn_bf, w1_bf + oW, b1 + i * D_FF, nullptr, nullptr, h_bf,
                                      ROWS, D_FF, D_MODEL, 1);      // relu
    gemm_wmma<<<gD, 256, 0, stream>>>(h_bf, w2_bf + oW, b2 + i * D_MODEL, x, x, nullptr,
                                      ROWS, D_MODEL, D_FF, 0);      // x += FFN
  }
  layernorm_k<<<ROWS, 256, 0, stream>>>(x, lnfa, lnfb, nullptr, (float*)d_out);
}
